// CIFAR10_Monarch_MLP2_30709016166962
// MI455X (gfx1250) — matrix operand order repaired and checked by reading, not compiled, not run on hardware
//
#include <hip/hip_runtime.h>
#include <hip/hip_bf16.h>

// ---------------------------------------------------------------------------
// Monarch MLP forward for MI455X (gfx1250), wave32 + WMMA f16 (f32 accumulate)
// f16 activation tiles staged to LDS via the Tensor Data Mover (TDM);
// weights converted f32->f16 while staging; padded-stride intermediates.
// ---------------------------------------------------------------------------

typedef _Float16 f16;
typedef __attribute__((ext_vector_type(16))) _Float16 v16h;
typedef __attribute__((ext_vector_type(8)))  _Float16 v8h;
typedef __attribute__((ext_vector_type(8)))  float    v8f;
typedef unsigned int u32x4 __attribute__((ext_vector_type(4)));
typedef int          i32x8 __attribute__((ext_vector_type(8)));
typedef int          i32x4 __attribute__((ext_vector_type(4)));

#if __has_builtin(__builtin_amdgcn_tensor_load_to_lds) && \
    __has_builtin(__builtin_amdgcn_s_wait_tensorcnt)
#define MONARCH_TDM 1
#else
#define MONARCH_TDM 0
#endif

static constexpr int BATCH = 16384;
static constexpr int MT = 64;    // workgroup M tile
static constexpr int NT = 128;   // workgroup N tile
static constexpr int KT = 32;    // K step per WMMA
static constexpr int LDSP = 40;  // padded LDS row (halves): 80B stride, 16B aligned

// Block-diagonal GEMM:  C[b, blk, n] = sum_k A[b, blk*blkStride + k] * W[blk, n, k]
//   A: (M, ldA) row-major, element type AT (float or _Float16); block k at col blk*blkStride
//   W: (4, N, K) row-major fp32, converted f32->f16 while staging to LDS
//   PERM=true : scatter f16 at (c&3)*outStride + (c>>2), c = blk*N+n  (monarch shuffle)
//   PERM=false: add bias[blk*N+n], optional ReLU, store f16 at blk*outStride + n
template <typename AT, bool PERM, bool RELU>
__global__ __launch_bounds__(256) void monarch_gemm(
    const AT* __restrict__ A, const float* __restrict__ W,
    const float* __restrict__ bias, f16* __restrict__ Cout,
    int N, int K, int ldA, int blkStride, int outStride) {
  __shared__ alignas(16) f16 As[2][MT][LDSP];
  __shared__ alignas(16) f16 Bs[2][NT][LDSP];

  const int tid  = threadIdx.x;
  const int blk  = blockIdx.z;              // block-diagonal block 0..3
  const int m0   = blockIdx.y * MT;
  const int n0   = blockIdx.x * NT;
  const int lane = tid & 31;
  const int wave = tid >> 5;                // 0..7
  const int wm   = (wave >> 2) * 32;        // wave M offset in tile
  const int wn   = (wave & 3) * 32;         // wave N offset in tile
  const int l16  = lane & 15;
  const int lh   = lane >> 4;               // lane half (0/1)

#if MONARCH_TDM
  constexpr bool kTdmA = (sizeof(AT) == 2);
#else
  constexpr bool kTdmA = false;
#endif

  // ---- staging thread mapping (fixed per thread) ----
  const int arow = tid >> 2;                // 0..63
  const int acol = (tid & 3) << 3;          // 0,8,16,24
  const int brow = tid >> 1;                // 0..127
  const int bcol = (tid & 1) << 4;          // 0,16
  const int gnB  = n0 + brow;
  const AT*    Abase = A + (size_t)(m0 + arow) * ldA + (size_t)blk * blkStride + acol;
  const float* Wbase = W + ((size_t)blk * N + (size_t)(gnB < N ? gnB : 0)) * K + bcol;

#if MONARCH_TDM
  // TDM: DMA one 64 x 32-half f16 tile (row stride ldA) into As[st] with
  // 16B LDS row padding (64B data + 16B pad = 80B = LDSP halves).
  // D# per CDNA5 ISA; 2D tensor (groups 2/3 = 0).
  auto tdm_load_A = [&](int st, int k0) {
    const unsigned ldsA = (unsigned)(size_t)(&As[st][0][0]);
    const unsigned long long ga =
        (unsigned long long)(size_t)((const char*)A +
            ((size_t)m0 * ldA + (size_t)blk * blkStride + k0) * sizeof(AT));
    const int rem = K - k0;  // OOB columns (>= rem) read as zero -> K-tail pad
    u32x4 g0;
    g0[0] = 1u;                                   // count=1, user-mode load
    g0[1] = ldsA;                                 // lds_addr
    g0[2] = (unsigned)ga;                         // global_addr[31:0]
    g0[3] = (unsigned)((ga >> 32) & 0x01FFFFFFu)  // global_addr[56:32]
            | (2u << 30);                         // type=2 ("image")
    i32x8 g1;
    g1[0] = (1 << 16)      // data_size = 2 bytes
          | (1 << 20)      // pad_enable
          | (3 << 22)      // pad_interval: 16 DWORDs (64B) between pads
          | (3 << 25);     // pad_amount: 4 DWORDs (16B)
    g1[1] = (rem & 0xFFFF) << 16;                        // tensor_dim0[15:0]
    g1[2] = ((rem >> 16) & 0xFFFF) | (MT << 16);         // dim0[31:16] | tensor_dim1[15:0]
    g1[3] = (KT << 16);                                  // dim1[31:16]=0 | tile_dim0
    g1[4] = MT;                                          // tile_dim1 | tile_dim2=0
    g1[5] = ldA;                                         // tensor_dim0_stride[31:0]
    g1[6] = 0;                                           // stride0[47:32] | stride1 lo
    g1[7] = 0;
    const i32x4 z4 = {0, 0, 0, 0};
    const i32x8 z8 = {0, 0, 0, 0, 0, 0, 0, 0};
    __builtin_amdgcn_tensor_load_to_lds(g0, g1, z4, z4, z8, 0);
  };
#endif

  auto load_stage = [&](int st, int k0) {
    const bool fullk = (k0 + KT) <= K;      // uniform: all tiles but the last
    // ---------- A tile: 64 x 32 halves ----------
    if constexpr (kTdmA) {
#if MONARCH_TDM
      if (wave == 0) tdm_load_A(st, k0);    // single DMA per workgroup stage
#endif
    } else {
      f16* dst = &As[st][arow][acol];
      const AT* Ap = Abase + k0;
      if (fullk) {
        if constexpr (sizeof(AT) == 4) {
          const float4 u0 = *(const float4*)(Ap);
          const float4 u1 = *(const float4*)(Ap + 4);
          v8h h;
          h[0] = (f16)u0.x; h[1] = (f16)u0.y; h[2] = (f16)u0.z; h[3] = (f16)u0.w;
          h[4] = (f16)u1.x; h[5] = (f16)u1.y; h[6] = (f16)u1.z; h[7] = (f16)u1.w;
          *(v8h*)dst = h;
        } else {
          *(v8h*)dst = *(const v8h*)Ap;
        }
      } else {
#pragma unroll
        for (int j = 0; j < 8; ++j) {
          const int k = k0 + acol + j;
          dst[j] = (k < K) ? (f16)(float)Ap[j] : (f16)0.f;
        }
      }
    }
    // ---------- B tile: 128 x 32, 16 floats/thread, f32->f16 convert ----------
    {
      f16* dst = &Bs[st][brow][bcol];
      if (gnB < N) {
        const float* Wp = Wbase + k0;
        if (fullk && (K & 1) == 0) {        // float2-aligned (K even for all full tiles)
          v8h h0, h1;
#pragma unroll
          for (int j = 0; j < 4; ++j) {
            const float2 u = *(const float2*)(Wp + 2 * j);
            h0[2 * j] = (f16)u.x; h0[2 * j + 1] = (f16)u.y;
          }
#pragma unroll
          for (int j = 0; j < 4; ++j) {
            const float2 u = *(const float2*)(Wp + 8 + 2 * j);
            h1[2 * j] = (f16)u.x; h1[2 * j + 1] = (f16)u.y;
          }
          *(v8h*)dst = h0;
          *(v8h*)(dst + 8) = h1;
        } else {
#pragma unroll
          for (int j = 0; j < 16; ++j) {
            const int k = k0 + bcol + j;
            dst[j] = (k < K) ? (f16)Wp[j] : (f16)0.f;
          }
        }
      } else {
        const v8h z = (v8h)(f16)0.f;
        *(v8h*)dst = z;
        *(v8h*)(dst + 8) = z;
      }
    }
  };

  auto tdm_fence = [&]() {
#if MONARCH_TDM
    if constexpr (kTdmA) {
      if (wave == 0) __builtin_amdgcn_s_wait_tensorcnt(0);
    }
#endif
  };

  v8f acc[2][2];
#pragma unroll
  for (int mi = 0; mi < 2; ++mi)
#pragma unroll
    for (int ni = 0; ni < 2; ++ni) acc[mi][ni] = (v8f)0.f;

  const int nk = (K + KT - 1) / KT;
  load_stage(0, 0);
  tdm_fence();
  __syncthreads();

  for (int kt = 0; kt < nk; ++kt) {
    const int st = kt & 1;
    if (kt + 1 < nk) load_stage(st ^ 1, (kt + 1) * KT);

    // Fragment loads per ISA VGPR layouts (wave32).
    v16h af[2], bf[2];
#pragma unroll
    for (int mi = 0; mi < 2; ++mi) {
      // A 16x32: lane -> M=l16; half h -> K = (h%8) + 16*(h/8) + 8*lh
      const f16* p = &As[st][wm + mi * 16 + l16][lh * 8];
      v8h lo = *(const v8h*)(p);
      v8h hi = *(const v8h*)(p + 16);
      af[mi] = __builtin_shufflevector(lo, hi, 0, 1, 2, 3, 4, 5, 6, 7, 8, 9,
                                       10, 11, 12, 13, 14, 15);
    }
#pragma unroll
    for (int ni = 0; ni < 2; ++ni) {
      // B 32x16: lane -> N=l16; half h -> K = (h%8) + 16*(h/8) + 8*lh
      const f16* p = &Bs[st][wn + ni * 16 + l16][lh * 8];
      v8h lo = *(const v8h*)(p);
      v8h hi = *(const v8h*)(p + 16);
      bf[ni] = __builtin_shufflevector(lo, hi, 0, 1, 2, 3, 4, 5, 6, 7, 8, 9,
                                       10, 11, 12, 13, 14, 15);
    }
#pragma unroll
    for (int mi = 0; mi < 2; ++mi)
#pragma unroll
      for (int ni = 0; ni < 2; ++ni)
        acc[mi][ni] = __builtin_amdgcn_wmma_f32_16x16x32_f16(
            false, af[mi], false, bf[ni], (short)0, acc[mi][ni], false, false);

    tdm_fence();  // TDM for next stage must land before the barrier publishes it
    __syncthreads();
  }

  // Epilogue. C/D layout: lane -> N=l16; vgpr j -> M = j + 8*lh.
  const size_t ldC = 4 * (size_t)outStride;
#pragma unroll
  for (int mi = 0; mi < 2; ++mi) {
#pragma unroll
    for (int ni = 0; ni < 2; ++ni) {
      const int gn = n0 + wn + ni * 16 + l16;
      if (gn >= N) continue;
      const int c = blk * N + gn;  // logical output column (k*q + q_idx)
      int oc;
      float bv = 0.f;
      if constexpr (PERM) {
        oc = (c & 3) * outStride + (c >> 2);  // monarch shuffle, padded block stride
      } else {
        oc = blk * outStride + gn;
        bv = bias[c];
      }
      f16* crow = Cout + (size_t)(m0 + wm + mi * 16 + lh * 8) * ldC + oc;
#pragma unroll
      for (int j = 0; j < 8; ++j) {
        float v = acc[mi][ni][j];
        if constexpr (!PERM) {
          v += bv;
          if constexpr (RELU) v = fmaxf(v, 0.f);
        }
        crow[(size_t)j * ldC] = (f16)v;
      }
    }
  }
}

// Layer 4 (4x(3x25), shuffle, 4x(3x3), truncate to 10, bias, log_softmax).
// H rows have padded layout: block k at column k*32, 25 valid, row width 128.
__global__ __launch_bounds__(256) void monarch_tail(
    const f16* __restrict__ H, const float* __restrict__ w1,
    const float* __restrict__ w2, const float* __restrict__ b,
    float* __restrict__ out, int M) {
  const int i = blockIdx.x * blockDim.x + threadIdx.x;
  if (i >= M) return;
  const f16* Hrow = H + (size_t)i * 128;
  float o1[12];
#pragma unroll
  for (int k = 0; k < 4; ++k)
#pragma unroll
    for (int q = 0; q < 3; ++q) {
      float s = 0.f;
#pragma unroll
      for (int p = 0; p < 25; ++p)
        s += w1[(k * 3 + q) * 25 + p] * (float)Hrow[k * 32 + p];
      o1[k * 3 + q] = s;
    }
  float lg[10];
#pragma unroll
  for (int j = 0; j < 10; ++j) {
    const int l = j / 3, si = j % 3;
    float s = b[j];
#pragma unroll
    for (int r = 0; r < 3; ++r) s += w2[l * 9 + si * 3 + r] * o1[r * 4 + l];
    lg[j] = s;
  }
  float m = lg[0];
#pragma unroll
  for (int j = 1; j < 10; ++j) m = fmaxf(m, lg[j]);
  float sum = 0.f;
#pragma unroll
  for (int j = 0; j < 10; ++j) sum += expf(lg[j] - m);
  const float ls = logf(sum);
#pragma unroll
  for (int j = 0; j < 10; ++j) out[(size_t)i * 10 + j] = lg[j] - m - ls;
}

static inline int cdiv(int a, int b) { return (a + b - 1) / b; }

extern "C" void kernel_launch(void* const* d_in, const int* in_sizes, int n_in,
                              void* d_out, int out_size, void* d_ws,
                              size_t ws_size, hipStream_t stream) {
  (void)in_sizes; (void)n_in; (void)out_size; (void)ws_size;
  const float* x    = (const float*)d_in[0];
  const float* w1_1 = (const float*)d_in[1];
  const float* w2_1 = (const float*)d_in[2];
  const float* b1   = (const float*)d_in[3];
  const float* w1_2 = (const float*)d_in[4];
  const float* w2_2 = (const float*)d_in[5];
  const float* b2   = (const float*)d_in[6];
  const float* w1_3 = (const float*)d_in[7];
  const float* w2_3 = (const float*)d_in[8];
  const float* b3   = (const float*)d_in[9];
  const float* w1_4 = (const float*)d_in[10];
  const float* w2_4 = (const float*)d_in[11];
  const float* b4   = (const float*)d_in[12];

  // two ping-pong f16 activation buffers, max padded row = 4*752 = 3008 halves
  const size_t BUF = (size_t)BATCH * 3072 * sizeof(f16);
  f16* P  = (f16*)d_ws;                      // permuted intermediates
  f16* Ab = (f16*)((char*)d_ws + BUF);       // post-layer activations

  const dim3 blk(256);
  const int mg = cdiv(BATCH, MT);            // 256 M tiles

  // Layer 1: x(f32, ld 3072, bs 768) -> P(perm, stride 752) -> Ab(relu, stride 752)
  monarch_gemm<float, true, false><<<dim3(cdiv(750, NT), mg, 4), blk, 0, stream>>>(
      x, w1_1, nullptr, P, 750, 768, 3072, 768, 752);
  monarch_gemm<f16, false, true><<<dim3(cdiv(750, NT), mg, 4), blk, 0, stream>>>(
      P, w2_1, b1, Ab, 750, 750, 3008, 752, 752);

  // Layer 2: Ab(ld 3008, bs 752) -> P(stride 256) -> Ab(stride 256)
  monarch_gemm<f16, true, false><<<dim3(cdiv(250, NT), mg, 4), blk, 0, stream>>>(
      Ab, w1_2, nullptr, P, 250, 750, 3008, 752, 256);
  monarch_gemm<f16, false, true><<<dim3(cdiv(250, NT), mg, 4), blk, 0, stream>>>(
      P, w2_2, b2, Ab, 250, 250, 1024, 256, 256);

  // Layer 3: Ab(ld 1024, bs 256) -> P(stride 32) -> Ab(stride 32)
  monarch_gemm<f16, true, false><<<dim3(1, mg, 4), blk, 0, stream>>>(
      Ab, w1_3, nullptr, P, 25, 250, 1024, 256, 32);
  monarch_gemm<f16, false, true><<<dim3(1, mg, 4), blk, 0, stream>>>(
      P, w2_3, b3, Ab, 25, 25, 128, 32, 32);

  // Layer 4 + log_softmax -> d_out (16384 x 10, f32)
  monarch_tail<<<cdiv(BATCH, 256), blk, 0, stream>>>(
      Ab, w1_4, w2_4, b4, (float*)d_out, BATCH);
}
